// LGEB_87351044866445
// MI455X (gfx1250) — matrix operand order repaired and checked by reading, not compiled, not run on hardware
//
#include <hip/hip_runtime.h>
#include <hip/hip_bf16.h>

// ---------------- types ----------------
typedef float  v8f   __attribute__((ext_vector_type(8)));
typedef __bf16 v16bf __attribute__((ext_vector_type(16)));

// ---------------- problem constants ----------------
#define Nn   50000
#define Ee   1600000
#define NH   72
#define NA   8
#define K1P  160   // padded K for 146 (edge feat) / 152 (node feat)
#define K2P  96    // padded K for 72
#define NP   80    // padded N for 72 (5 tiles of 16)

// ---------------- workspace byte offsets ----------------
#define WS_WE1T   0u          // 80*160*2 = 25600
#define WS_WH1T   25600u
#define WS_WE2T   51200u      // 80*96*2 = 15360
#define WS_WX1T   66560u
#define WS_WH2T   81920u
#define WS_HBF    97280u      // 50000*72*2 = 7,200,000
#define WS_Z1     7297280u    // 1,600,000*72*2 = 230,400,000 (bf16)
#define WS_Z3     237697280u  // 50000*72*4
#define WS_AGG    252097280u  // 50000*72*4
#define WS_AGGX   266497280u  // 50000*4*4
#define WS_CNT    267297280u  // 50000*4
#define WS_STATS  267497280u  // 576 floats
// zero region: AGG..STATS end = 15,402,304 bytes = 3,850,576 floats

// ---------------- async global->LDS staging (CDNA5 path w/ fallback) ----------------
#if defined(__AMDGCN__) && __has_builtin(__builtin_amdgcn_global_load_async_to_lds_b128) && \
    __has_builtin(__builtin_amdgcn_s_wait_asynccnt)
#define ASYNC_LDS 1
#endif

static __device__ inline void stage_weights(__bf16* dst, const __bf16* src, int nbytes,
                                            int tid, int nthr) {
#ifdef ASYNC_LDS
  typedef int v4i_t __attribute__((vector_size(16)));
  typedef __attribute__((address_space(1))) v4i_t gbl_v4i;
  typedef __attribute__((address_space(3))) v4i_t lds_v4i;
  gbl_v4i* s = (gbl_v4i*)src;
  lds_v4i* d = (lds_v4i*)dst;
  for (int t = tid; t < nbytes / 16; t += nthr)
    __builtin_amdgcn_global_load_async_to_lds_b128(s + t, d + t, 0, 0);
  __builtin_amdgcn_s_wait_asynccnt(0);
#else
  for (int t = tid; t < nbytes / 16; t += nthr)
    ((uint4*)dst)[t] = ((const uint4*)src)[t];
#endif
}

// ---------------- WMMA fragment helpers ----------------
static __device__ inline v16bf load_a_frag(const __bf16* row, int koff) {
  // A 16x32 tile in this code's contiguous k order (k = 16h + s, same as B): 16 K values per lane (32B)
  union { uint4 u[2]; v16bf v; } t;
  t.u[0] = *(const uint4*)(row + koff);
  t.u[1] = *(const uint4*)(row + koff + 8);
  return t.v;
}

static __device__ inline v16bf load_b_frag(const __bf16* wt, int n, int kp, int kb) {
  // Weights stored N-major ([NP][kp]); lane's 16 K values are contiguous (32B)
  return *(const v16bf*)(wt + (size_t)n * kp + kb);
}

static __device__ inline float psif(float v) {
  return copysignf(logf(fabsf(v) + 1.0f), v);
}

// ---------------- prep kernels ----------------
__global__ void k_prep_h(const float* __restrict__ h, __bf16* __restrict__ hbf, int n) {
  int i = blockIdx.x * 256 + threadIdx.x;
  if (i < n) hbf[i] = (__bf16)h[i];
}

__global__ void k_prep_w(const float* __restrict__ We1, const float* __restrict__ Wh1,
                         const float* __restrict__ We2, const float* __restrict__ Wx1,
                         const float* __restrict__ Wh2,
                         __bf16* __restrict__ We1T, __bf16* __restrict__ Wh1T,
                         __bf16* __restrict__ We2T, __bf16* __restrict__ Wx1T,
                         __bf16* __restrict__ Wh2T) {
  int idx = blockIdx.x * 256 + threadIdx.x;
  if (idx < 12800) {                       // We1T [80][160] <- We1 [146][72]
    int n = idx / K1P, k = idx % K1P;
    We1T[idx] = (__bf16)((n < NH && k < 146) ? We1[k * NH + n] : 0.0f);
  } else if (idx < 25600) {                // Wh1T [80][160] <- Wh1 [152][72]
    int l = idx - 12800; int n = l / K1P, k = l % K1P;
    Wh1T[l] = (__bf16)((n < NH && k < 152) ? Wh1[k * NH + n] : 0.0f);
  } else if (idx < 33280) {                // We2T [80][96] <- We2 [72][72]
    int l = idx - 25600; int n = l / K2P, k = l % K2P;
    We2T[l] = (__bf16)((n < NH && k < NH) ? We2[k * NH + n] : 0.0f);
  } else if (idx < 40960) {                // Wx1T
    int l = idx - 33280; int n = l / K2P, k = l % K2P;
    Wx1T[l] = (__bf16)((n < NH && k < NH) ? Wx1[k * NH + n] : 0.0f);
  } else if (idx < 48640) {                // Wh2T
    int l = idx - 40960; int n = l / K2P, k = l % K2P;
    Wh2T[l] = (__bf16)((n < NH && k < NH) ? Wh2[k * NH + n] : 0.0f);
  }
}

__global__ void k_zero(float4* __restrict__ p, int n4) {
  int i = blockIdx.x * 256 + threadIdx.x;
  if (i < n4) p[i] = float4{0.f, 0.f, 0.f, 0.f};
}

// ---------------- BN finalize ----------------
__global__ void k_bn_finalize(const float* __restrict__ sum, const float* __restrict__ sumsq,
                              const float* __restrict__ g, const float* __restrict__ b,
                              float* __restrict__ scale, float* __restrict__ shift,
                              float count) {
  int n = threadIdx.x;
  if (n < NH) {
    float mean = sum[n] / count;
    float var  = sumsq[n] / count - mean * mean;
    float sc   = g[n] * rsqrtf(var + 1e-5f);
    scale[n] = sc;
    shift[n] = b[n] - mean * sc;
  }
}

// ---------------- edge GEMM1: z1 = feat @ We1 (+BN stats) ----------------
// 3 waves/block, 32 rows (2 M-tiles) per wave: B-fragments reused across 2 A-tiles.
__global__ __launch_bounds__(96) void k_edge_gemm1(
    const __bf16* __restrict__ hbf, const float* __restrict__ x,
    const int* __restrict__ edges, const __bf16* __restrict__ We1T,
    __bf16* __restrict__ z1, float* __restrict__ sum_e, float* __restrict__ sumsq_e) {
  __shared__ __align__(32) __bf16 sW[NP * K1P];          // 25600 B
  __shared__ __align__(32) __bf16 sA[3][32 * K1P];       // 3 x 10240 B
  const int tid = threadIdx.x, lane = tid & 31, wave = tid >> 5;

  stage_weights(sW, We1T, NP * K1P * 2, tid, 96);

  const int ebase = blockIdx.x * 96 + wave * 32;
  // each lane stages one full feat row (row = lane)
  {
    const int e  = ebase + lane;
    const int ec = (e < Ee) ? e : 0;
    const int iv = edges[ec], jv = edges[Ee + ec];
    __bf16* arow = &sA[wave][lane * K1P];
    const uint4* si = (const uint4*)(hbf + (size_t)iv * NH);
    const uint4* sj = (const uint4*)(hbf + (size_t)jv * NH);
    uint4* di = (uint4*)arow;
    uint4* dj = (uint4*)(arow + NH);
#pragma unroll
    for (int t = 0; t < 9; ++t) { di[t] = si[t]; dj[t] = sj[t]; }
    float4 xi = *(const float4*)(x + iv * 4);
    float4 xj = *(const float4*)(x + jv * 4);
    float d0 = xi.x - xj.x, d1 = xi.y - xj.y, d2 = xi.z - xj.z, d3 = xi.w - xj.w;
    float nrm = d0 * d0 - d1 * d1 - d2 * d2 - d3 * d3;
    float dt  = xi.x * xj.x - xi.y * xj.y - xi.z * xj.z - xi.w * xj.w;
    arow[144] = (__bf16)psif(nrm);
    arow[145] = (__bf16)psif(dt);
    uint* pz = (uint*)(arow + 146);
#pragma unroll
    for (int t = 0; t < 7; ++t) pz[t] = 0u;
  }
  __syncthreads();

  v8f acc[2][5] = {};
  const __bf16* A0 = &sA[wave][(lane & 15) * K1P];
  const __bf16* A1 = &sA[wave][(16 + (lane & 15)) * K1P];
#pragma unroll
  for (int kc = 0; kc < 5; ++kc) {
    const int koff = kc * 32 + ((lane & 16) ? 16 : 0);
    v16bf a0 = load_a_frag(A0, koff);
    v16bf a1 = load_a_frag(A1, koff);
    const int kb = kc * 32 + ((lane & 16) ? 16 : 0);
#pragma unroll
    for (int t = 0; t < 5; ++t) {
      v16bf b = load_b_frag(sW, t * 16 + (lane & 15), K1P, kb);
      acc[0][t] = __builtin_amdgcn_wmma_f32_16x16x32_bf16(false, a0, false, b, (short)0,
                                                          acc[0][t], false, false);
      acc[1][t] = __builtin_amdgcn_wmma_f32_16x16x32_bf16(false, a1, false, b, (short)0,
                                                          acc[1][t], false, false);
    }
  }

  const int mofs = (lane & 16) ? 8 : 0;
#pragma unroll
  for (int s = 0; s < 2; ++s) {
    const int rowbase = ebase + s * 16 + mofs;
#pragma unroll
    for (int t = 0; t < 5; ++t) {
      const int n = t * 16 + (lane & 15);
      if (n < NH) {
        float sm = 0.f, sq = 0.f;
#pragma unroll
        for (int rr = 0; rr < 8; ++rr) {
          const int row = rowbase + rr;
          if (row < Ee) {
            float v = acc[s][t][rr];
            z1[(size_t)row * NH + n] = (__bf16)v;
            sm += v; sq += v * v;
          }
        }
        atomicAdd(&sum_e[n], sm);
        atomicAdd(&sumsq_e[n], sq);
      }
    }
  }
}

// ---------------- fused edge tail: BN->ReLU->GEMM2->gate->m/agg->GEMM3->px->x scatter --------
__global__ __launch_bounds__(64) void k_edge_part2(
    const __bf16* __restrict__ z1, const float* __restrict__ x,
    const int* __restrict__ edges,
    const __bf16* __restrict__ We2T, const __bf16* __restrict__ Wx1T,
    const float* __restrict__ scale_e, const float* __restrict__ shift_e,
    const float* __restrict__ be2, const float* __restrict__ Wm,
    const float* __restrict__ bm, const float* __restrict__ bx1,
    const float* __restrict__ Wx2,
    float* __restrict__ out_m, float* __restrict__ agg,
    float* __restrict__ aggx, float* __restrict__ cnt) {
  __shared__ __align__(32) __bf16 sW2[NP * K2P];
  __shared__ __align__(32) __bf16 sWx[NP * K2P];
  __shared__ __align__(32) __bf16 sA[2][16 * K2P];
  __shared__ float sF[2][16 * NP];
  __shared__ float sScale[NH], sShift[NH], sBe2[NP], sBx1[NP], sWmv[NH], sWx2v[NH];
  __shared__ float sRowW[2][16];
  __shared__ int sI[2][16], sJ[2][16];
  const int tid = threadIdx.x, lane = tid & 31, wave = tid >> 5;

  stage_weights(sW2, We2T, NP * K2P * 2, tid, 64);
  stage_weights(sWx, Wx1T, NP * K2P * 2, tid, 64);
  if (tid < NH) {
    sScale[tid] = scale_e[tid]; sShift[tid] = shift_e[tid];
    sBe2[tid] = be2[tid]; sBx1[tid] = bx1[tid];
    sWmv[tid] = Wm[tid]; sWx2v[tid] = Wx2[tid];
  } else if (tid < NP) { sBe2[tid] = 0.f; sBx1[tid] = 0.f; }

  const int eb = blockIdx.x * 32 + wave * 16;
  const int r = lane & 15, half = lane >> 4;
  if (half == 0) { sI[wave][r] = edges[eb + r]; sJ[wave][r] = edges[Ee + eb + r]; }
  __syncthreads();

  // stage A = relu(bn(z1))
  {
    const __bf16* zrow = z1 + (size_t)(eb + r) * NH;
    __bf16* arow = &sA[wave][r * K2P];
#pragma unroll
    for (int c0 = 0; c0 < 36; ++c0) {
      int c = half * 36 + c0;
      float v = (float)zrow[c];
      v = fmaxf(v * sScale[c] + sShift[c], 0.f);
      arow[c] = (__bf16)v;
    }
    if (half) {
#pragma unroll
      for (int c = NH; c < K2P; ++c) arow[c] = (__bf16)0.f;
    }
  }
  __syncthreads();

  const __bf16* Arow = &sA[wave][(lane & 15) * K2P];
  const int mofs = (lane & 16) ? 8 : 0;

  // GEMM2: e2 = relu(A @ We2 + be2)
  {
    v8f acc[5] = {};
#pragma unroll
    for (int kc = 0; kc < 3; ++kc) {
      const int koff = kc * 32 + ((lane & 16) ? 16 : 0);
      v16bf a = load_a_frag(Arow, koff);
      const int kb = kc * 32 + ((lane & 16) ? 16 : 0);
#pragma unroll
      for (int t = 0; t < 5; ++t) {
        v16bf b = load_b_frag(sW2, t * 16 + (lane & 15), K2P, kb);
        acc[t] = __builtin_amdgcn_wmma_f32_16x16x32_bf16(false, a, false, b, (short)0,
                                                         acc[t], false, false);
      }
    }
#pragma unroll
    for (int t = 0; t < 5; ++t) {
      const int n = t * 16 + (lane & 15);
#pragma unroll
      for (int rr = 0; rr < 8; ++rr)
        sF[wave][(rr + mofs) * NP + n] = fmaxf(acc[t][rr] + sBe2[n], 0.f);
    }
  }
  __syncthreads();

  // gate w = sigmoid(e2 @ Wm + bm)
  if (half == 0) {
    float dot = bm[0];
#pragma unroll 8
    for (int c = 0; c < NH; ++c) dot += sF[wave][r * NP + c] * sWmv[c];
    sRowW[wave][r] = 1.f / (1.f + expf(-dot));
  }
  __syncthreads();

  // m = e2*w : emit to out, scatter to agg, restage as bf16 A tile
  for (int idx = lane; idx < 16 * NH; idx += 32) {
    int rr = idx / NH, c = idx - rr * NH;
    float mv = sF[wave][rr * NP + c] * sRowW[wave][rr];
    out_m[(size_t)(eb + rr) * NH + c] = mv;
    atomicAdd(&agg[(size_t)sI[wave][rr] * NH + c], mv);
    sA[wave][rr * K2P + c] = (__bf16)mv;
  }
  for (int idx = lane; idx < 16 * (K2P - NH); idx += 32) {
    int rr = idx / (K2P - NH), c = NH + (idx - rr * (K2P - NH));
    sA[wave][rr * K2P + c] = (__bf16)0.f;
  }
  __syncthreads();

  // GEMM3: p = relu(m @ Wx1 + bx1)
  {
    v8f acc[5] = {};
#pragma unroll
    for (int kc = 0; kc < 3; ++kc) {
      const int koff = kc * 32 + ((lane & 16) ? 16 : 0);
      v16bf a = load_a_frag(Arow, koff);
      const int kb = kc * 32 + ((lane & 16) ? 16 : 0);
#pragma unroll
      for (int t = 0; t < 5; ++t) {
        v16bf b = load_b_frag(sWx, t * 16 + (lane & 15), K2P, kb);
        acc[t] = __builtin_amdgcn_wmma_f32_16x16x32_bf16(false, a, false, b, (short)0,
                                                         acc[t], false, false);
      }
    }
#pragma unroll
    for (int t = 0; t < 5; ++t) {
      const int n = t * 16 + (lane & 15);
#pragma unroll
      for (int rr = 0; rr < 8; ++rr)
        sF[wave][(rr + mofs) * NP + n] = fmaxf(acc[t][rr] + sBx1[n], 0.f);
    }
  }
  __syncthreads();

  // px = p @ Wx2 ; trans = clip(x_diff*px); scatter
  if (half == 0) {
    float px = 0.f;
#pragma unroll 8
    for (int c = 0; c < NH; ++c) px += sF[wave][r * NP + c] * sWx2v[c];
    const int iv = sI[wave][r], jv = sJ[wave][r];
    float4 xi = *(const float4*)(x + iv * 4);
    float4 xj = *(const float4*)(x + jv * 4);
    float t0 = fminf(fmaxf((xi.x - xj.x) * px, -100.f), 100.f);
    float t1 = fminf(fmaxf((xi.y - xj.y) * px, -100.f), 100.f);
    float t2 = fminf(fmaxf((xi.z - xj.z) * px, -100.f), 100.f);
    float t3 = fminf(fmaxf((xi.w - xj.w) * px, -100.f), 100.f);
    atomicAdd(&aggx[iv * 4 + 0], t0);
    atomicAdd(&aggx[iv * 4 + 1], t1);
    atomicAdd(&aggx[iv * 4 + 2], t2);
    atomicAdd(&aggx[iv * 4 + 3], t3);
    atomicAdd(&cnt[iv], 1.f);
  }
}

// ---------------- node GEMM1: z3 = hcat @ Wh1 + bh1 (+BN stats) ----------------
__global__ __launch_bounds__(128) void k_node_gemm1(
    const float* __restrict__ h, const float* __restrict__ agg,
    const float* __restrict__ node_attr, const __bf16* __restrict__ Wh1T,
    const float* __restrict__ bh1, float* __restrict__ z3,
    float* __restrict__ sum_h, float* __restrict__ sumsq_h) {
  __shared__ __align__(32) __bf16 sW[NP * K1P];
  __shared__ __align__(32) __bf16 sA[4][16 * K1P];
  __shared__ float sBh1[NP];
  const int tid = threadIdx.x, lane = tid & 31, wave = tid >> 5;

  stage_weights(sW, Wh1T, NP * K1P * 2, tid, 128);
  if (tid < NH) sBh1[tid] = bh1[tid];
  else if (tid < NP) sBh1[tid] = 0.f;

  const int nb = blockIdx.x * 64 + wave * 16;
  const int r = lane & 15, half = lane >> 4;
  const int v = nb + r;
  const int vc = (v < Nn) ? v : (Nn - 1);
  __bf16* arow = &sA[wave][r * K1P];
  if (half == 0) {
    const float* hr = h + (size_t)vc * NH;
#pragma unroll 8
    for (int c = 0; c < NH; ++c) arow[c] = (__bf16)hr[c];
  } else {
    const float* ar = agg + (size_t)vc * NH;
#pragma unroll 8
    for (int c = 0; c < NH; ++c) arow[NH + c] = (__bf16)ar[c];
    const float* na = node_attr + (size_t)vc * NA;
#pragma unroll
    for (int c = 0; c < NA; ++c) arow[144 + c] = (__bf16)na[c];
#pragma unroll
    for (int c = 152; c < K1P; ++c) arow[c] = (__bf16)0.f;
  }
  __syncthreads();

  v8f acc[5] = {};
  const __bf16* Arow = &sA[wave][(lane & 15) * K1P];
#pragma unroll
  for (int kc = 0; kc < 5; ++kc) {
    const int koff = kc * 32 + ((lane & 16) ? 16 : 0);
    v16bf a = load_a_frag(Arow, koff);
    const int kb = kc * 32 + ((lane & 16) ? 16 : 0);
#pragma unroll
    for (int t = 0; t < 5; ++t) {
      v16bf b = load_b_frag(sW, t * 16 + (lane & 15), K1P, kb);
      acc[t] = __builtin_amdgcn_wmma_f32_16x16x32_bf16(false, a, false, b, (short)0,
                                                       acc[t], false, false);
    }
  }

  const int mofs = (lane & 16) ? 8 : 0;
#pragma unroll
  for (int t = 0; t < 5; ++t) {
    const int n = t * 16 + (lane & 15);
    if (n < NH) {
      float s = 0.f, sq = 0.f;
#pragma unroll
      for (int rr = 0; rr < 8; ++rr) {
        const int row = nb + rr + mofs;
        if (row < Nn) {
          float val = acc[t][rr] + sBh1[n];
          z3[(size_t)row * NH + n] = val;
          s += val; sq += val * val;
        }
      }
      atomicAdd(&sum_h[n], s);
      atomicAdd(&sumsq_h[n], sq);
    }
  }
}

// ---------------- node final: h_out = h + relu(bn(z3)) @ Wh2 + bh2 ; x_out ----------------
__global__ __launch_bounds__(128) void k_node_final(
    const float* __restrict__ z3, const float* __restrict__ h,
    const float* __restrict__ x, const __bf16* __restrict__ Wh2T,
    const float* __restrict__ scale_h, const float* __restrict__ shift_h,
    const float* __restrict__ bh2, const float* __restrict__ aggx,
    const float* __restrict__ cnt, float* __restrict__ out_h,
    float* __restrict__ out_x) {
  __shared__ __align__(32) __bf16 sW[NP * K2P];
  __shared__ __align__(32) __bf16 sA[4][16 * K2P];
  __shared__ float sScale[NH], sShift[NH], sBh2[NP];
  const int tid = threadIdx.x, lane = tid & 31, wave = tid >> 5;

  stage_weights(sW, Wh2T, NP * K2P * 2, tid, 128);
  if (tid < NH) { sScale[tid] = scale_h[tid]; sShift[tid] = shift_h[tid]; sBh2[tid] = bh2[tid]; }
  else if (tid < NP) sBh2[tid] = 0.f;

  const int nb = blockIdx.x * 64 + wave * 16;
  const int r = lane & 15, half = lane >> 4;
  const int v = nb + r;
  const int vc = (v < Nn) ? v : (Nn - 1);
  __syncthreads();

  {
    const float* zrow = z3 + (size_t)vc * NH;
    __bf16* arow = &sA[wave][r * K2P];
#pragma unroll
    for (int c0 = 0; c0 < 36; ++c0) {
      int c = half * 36 + c0;
      float val = fmaxf(zrow[c] * sScale[c] + sShift[c], 0.f);
      arow[c] = (__bf16)val;
    }
    if (half) {
#pragma unroll
      for (int c = NH; c < K2P; ++c) arow[c] = (__bf16)0.f;
    }
  }
  __syncthreads();

  v8f acc[5] = {};
  const __bf16* Arow = &sA[wave][(lane & 15) * K2P];
#pragma unroll
  for (int kc = 0; kc < 3; ++kc) {
    const int koff = kc * 32 + ((lane & 16) ? 16 : 0);
    v16bf a = load_a_frag(Arow, koff);
    const int kb = kc * 32 + ((lane & 16) ? 16 : 0);
#pragma unroll
    for (int t = 0; t < 5; ++t) {
      v16bf b = load_b_frag(sW, t * 16 + (lane & 15), K2P, kb);
      acc[t] = __builtin_amdgcn_wmma_f32_16x16x32_bf16(false, a, false, b, (short)0,
                                                       acc[t], false, false);
    }
  }

  const int mofs = (lane & 16) ? 8 : 0;
#pragma unroll
  for (int t = 0; t < 5; ++t) {
    const int n = t * 16 + (lane & 15);
    if (n < NH) {
#pragma unroll
      for (int rr = 0; rr < 8; ++rr) {
        const int row = nb + rr + mofs;
        if (row < Nn)
          out_h[(size_t)row * NH + n] = h[(size_t)row * NH + n] + acc[t][rr] + sBh2[n];
      }
    }
  }
  if (half == 0 && v < Nn) {
    float d = fmaxf(cnt[v], 1.f);
#pragma unroll
    for (int k = 0; k < 4; ++k)
      out_x[v * 4 + k] = x[v * 4 + k] + aggx[v * 4 + k] / d;
  }
}

// ---------------- launch ----------------
extern "C" void kernel_launch(void* const* d_in, const int* in_sizes, int n_in,
                              void* d_out, int out_size, void* d_ws, size_t ws_size,
                              hipStream_t stream) {
  const float* h         = (const float*)d_in[0];
  const float* x         = (const float*)d_in[1];
  const float* node_attr = (const float*)d_in[2];
  const int*   edges     = (const int*)d_in[3];
  const float* We1  = (const float*)d_in[4];
  const float* bn_e_g = (const float*)d_in[5];
  const float* bn_e_b = (const float*)d_in[6];
  const float* We2  = (const float*)d_in[7];
  const float* be2  = (const float*)d_in[8];
  const float* Wh1  = (const float*)d_in[9];
  const float* bh1  = (const float*)d_in[10];
  const float* bn_h_g = (const float*)d_in[11];
  const float* bn_h_b = (const float*)d_in[12];
  const float* Wh2  = (const float*)d_in[13];
  const float* bh2  = (const float*)d_in[14];
  const float* Wx1  = (const float*)d_in[15];
  const float* bx1  = (const float*)d_in[16];
  const float* Wx2  = (const float*)d_in[17];
  const float* Wm   = (const float*)d_in[18];
  const float* bm   = (const float*)d_in[19];

  char* ws = (char*)d_ws;
  __bf16* We1T = (__bf16*)(ws + WS_WE1T);
  __bf16* Wh1T = (__bf16*)(ws + WS_WH1T);
  __bf16* We2T = (__bf16*)(ws + WS_WE2T);
  __bf16* Wx1T = (__bf16*)(ws + WS_WX1T);
  __bf16* Wh2T = (__bf16*)(ws + WS_WH2T);
  __bf16* hbf  = (__bf16*)(ws + WS_HBF);
  __bf16* z1   = (__bf16*)(ws + WS_Z1);
  float*  z3   = (float*)(ws + WS_Z3);
  float*  agg  = (float*)(ws + WS_AGG);
  float*  aggx = (float*)(ws + WS_AGGX);
  float*  cnt  = (float*)(ws + WS_CNT);
  float*  stats   = (float*)(ws + WS_STATS);
  float*  sum_e   = stats;
  float*  sumsq_e = stats + 72;
  float*  sum_h   = stats + 144;
  float*  sumsq_h = stats + 216;
  float*  scale_e = stats + 288;
  float*  shift_e = stats + 360;
  float*  scale_h = stats + 432;
  float*  shift_h = stats + 504;

  float* out_h = (float*)d_out;
  float* out_x = out_h + (size_t)Nn * NH;          // + 3,600,000
  float* out_m = out_x + (size_t)Nn * 4;           // + 200,000

  // prep + zero
  k_prep_h<<<(Nn * NH + 255) / 256, 256, 0, stream>>>(h, hbf, Nn * NH);
  k_prep_w<<<190, 256, 0, stream>>>(We1, Wh1, We2, Wx1, Wh2, We1T, Wh1T, We2T, Wx1T, Wh2T);
  {
    const int n4 = 3850576;  // (agg + aggx + cnt + stats) / float4
    k_zero<<<(n4 + 255) / 256, 256, 0, stream>>>((float4*)agg, n4);
  }
  // edge phase
  k_edge_gemm1<<<(Ee + 95) / 96, 96, 0, stream>>>(hbf, x, edges, We1T, z1, sum_e, sumsq_e);
  k_bn_finalize<<<1, 128, 0, stream>>>(sum_e, sumsq_e, bn_e_g, bn_e_b, scale_e, shift_e,
                                       (float)Ee);
  k_edge_part2<<<Ee / 32, 64, 0, stream>>>(z1, x, edges, We2T, Wx1T, scale_e, shift_e,
                                           be2, Wm, bm, bx1, Wx2, out_m, agg, aggx, cnt);
  // node phase
  k_node_gemm1<<<(Nn + 63) / 64, 128, 0, stream>>>(h, agg, node_attr, Wh1T, bh1, z3,
                                                   sum_h, sumsq_h);
  k_bn_finalize<<<1, 128, 0, stream>>>(sum_h, sumsq_h, bn_h_g, bn_h_b, scale_h, shift_h,
                                       (float)Nn);
  k_node_final<<<(Nn + 63) / 64, 128, 0, stream>>>(z3, h, x, Wh2T, scale_h, shift_h, bh2,
                                                   aggx, cnt, out_h, out_x);
}
